// SLAMNet_41661182771327
// MI455X (gfx1250) — matrix operand order repaired and checked by reading, not compiled, not run on hardware
//
#include <hip/hip_runtime.h>
#include <hip/hip_bf16.h>
#include <math.h>

// ---- problem constants (match reference) ----
#define N_NODES 16384
#define N_EDGES 262144
#define N_GRAPH 64
#define HDIM    128
#define NHEADS  4
#define CDIM    32
#define NLAYERS 3

typedef _Float16 v8h  __attribute__((ext_vector_type(8)));
typedef _Float16 v16h __attribute__((ext_vector_type(16)));
typedef float    v8f  __attribute__((ext_vector_type(8)));

#define ACT_NONE    0
#define ACT_RELU    1
#define ACT_GELU    2
#define ACT_SIGMOID 3

// ---- WMMA GEMM tiling ----
#define TM 128      // rows per block
#define TN 128      // cols per block
#define KCH 64      // K chunk staged in LDS
#define LDA 72      // padded half-stride (16B-aligned fragment loads)
#define LDB 72

// Generic fused GEMM: Y[M,Ntot] = epilogue( X[M,K] @ W[K,Ntot] )
// epilogue: v = acc + bias[col]; v += resid; act(v); v = v*scale[col]*bnMul + shift[col]
// XMODE==1: X is the virtual concat [hV[src[row]] | hE[row] | hV[dst[row]]], K=384.
template<int ACT, int XMODE>
__global__ __launch_bounds__(256)
void gemm_wmma_kernel(const float* __restrict__ X, int ldx,
                      const float* __restrict__ W, int ldw,
                      const float* __restrict__ bias,
                      const float* __restrict__ resid,
                      const float* __restrict__ scale,
                      const float* __restrict__ shift, float bnMul,
                      float* __restrict__ Y, int ldy,
                      int M, int K, int Ntot,
                      const int* __restrict__ gsrc, const int* __restrict__ gdst,
                      const float* __restrict__ gV, const float* __restrict__ gE)
{
    __shared__ _Float16 As[TM * LDA];
    __shared__ _Float16 Bs[TN * LDB];

    const int m0   = blockIdx.x * TM;
    const int n0   = blockIdx.y * TN;
    const int tid  = threadIdx.x;
    const int wave = tid >> 5;
    const int lane = tid & 31;
    const int r    = lane & 15;
    const int hl   = lane >> 4;

    v8f acc[8];
#pragma unroll
    for (int j = 0; j < 8; ++j) acc[j] = (v8f)0.0f;

    for (int kc = 0; kc < K; kc += KCH) {
        __syncthreads();
        // ---- stage A chunk (TM x KCH) as f16, row-major ----
#pragma unroll
        for (int i = 0; i < (TM * KCH) / 256; ++i) {
            int l   = i * 256 + tid;
            int row = l >> 6;          // KCH == 64
            int col = l & 63;
            int gm  = m0 + row;
            int gk  = kc + col;
            float v = 0.0f;
            if (gm < M && gk < K) {
                if (XMODE == 0) {
                    v = X[(size_t)gm * ldx + gk];
                } else {
                    if      (gk < HDIM)     v = gV[(size_t)gsrc[gm] * HDIM + gk];
                    else if (gk < 2 * HDIM) v = gE[(size_t)gm * HDIM + (gk - HDIM)];
                    else                    v = gV[(size_t)gdst[gm] * HDIM + (gk - 2 * HDIM)];
                }
            }
            As[row * LDA + col] = (_Float16)v;
        }
        // ---- stage B chunk transposed: Bs[n][k] (coalesced on n) ----
#pragma unroll
        for (int i = 0; i < (TN * KCH) / 256; ++i) {
            int l = i * 256 + tid;
            int k = l >> 7;            // TN == 128
            int n = l & 127;
            int gk = kc + k, gn = n0 + n;
            float v = 0.0f;
            if (gk < K && gn < Ntot) v = W[(size_t)gk * ldw + gn];
            Bs[n * LDB + k] = (_Float16)v;
        }
        __syncthreads();

#pragma unroll
        for (int ks = 0; ks < 2; ++ks) {
            // A fragment (16x32 f16): lane holds row wave*16+r,
            // halves: K = ks*32 + {hl*8..hl*8+7} U {16+hl*8..16+hl*8+7}
            const _Float16* ap = &As[(wave * 16 + r) * LDA + ks * 32 + hl * 8];
            v8h a0 = *(const v8h*)ap;
            v8h a1 = *(const v8h*)(ap + 16);
            v16h a = __builtin_shufflevector(a0, a1, 0,1,2,3,4,5,6,7,8,9,10,11,12,13,14,15);
#pragma unroll
            for (int jt = 0; jt < 8; ++jt) {
                // B fragment (32x16 f16): lane holds col jt*16+r,
                // halves: K = ks*32 + {hl*8..hl*8+7} U {16+hl*8..16+hl*8+7}
                const _Float16* bp = &Bs[(jt * 16 + r) * LDB + ks * 32 + hl * 8];
                v8h b0 = *(const v8h*)bp;
                v8h b1 = *(const v8h*)(bp + 16);
                v16h b = __builtin_shufflevector(b0, b1, 0,1,2,3,4,5,6,7,8,9,10,11,12,13,14,15);
                acc[jt] = __builtin_amdgcn_wmma_f32_16x16x32_f16(
                    false, a, false, b, (short)0, acc[jt], false, false);
            }
        }
    }

    // ---- epilogue: C/D layout => lane: N = jt*16 + r, M = wave*16 + hl*8 + g ----
#pragma unroll
    for (int jt = 0; jt < 8; ++jt) {
        int gn = n0 + jt * 16 + r;
        if (gn >= Ntot) continue;
        float bcol = bias ? bias[gn] : 0.0f;
        float sc   = scale ? scale[gn] * bnMul : 1.0f;
        float sh   = shift ? shift[gn] : 0.0f;
#pragma unroll
        for (int g = 0; g < 8; ++g) {
            int gm = m0 + wave * 16 + hl * 8 + g;
            if (gm >= M) continue;
            float v = acc[jt][g] + bcol;
            if (resid) v += resid[(size_t)gm * ldy + gn];
            if (ACT == ACT_RELU)         v = fmaxf(v, 0.0f);
            else if (ACT == ACT_GELU)    v = 0.5f * v * (1.0f + erff(v * 0.70710678118654752f));
            else if (ACT == ACT_SIGMOID) v = 1.0f / (1.0f + __expf(-v));
            if (scale) v = v * sc + sh;
            Y[(size_t)gm * ldy + gn] = v;
        }
    }
}

// ---- float atomic max via signed/unsigned bit trick (init with -inf) ----
__device__ __forceinline__ void atomicMaxF(float* a, float v) {
    if (v >= 0.0f) atomicMax((int*)a, __float_as_int(v));
    else           atomicMin((unsigned int*)a, (unsigned int)__float_as_int(v));
}

__global__ void fill_kernel(float* p, float v, int n) {
    int i = blockIdx.x * 256 + threadIdx.x;
    if (i < n) p[i] = v;
}

// logits[e,h] = dot(qh[dst], kh[src]+eh[e]) / sqrt(C); also running segment max.
__global__ void attn_logits_kernel(const float* __restrict__ qh, const float* __restrict__ kh,
                                   const float* __restrict__ eh,
                                   const int* __restrict__ src, const int* __restrict__ dst,
                                   float* __restrict__ logits, float* __restrict__ mbuf) {
    int idx = blockIdx.x * 256 + threadIdx.x;
    if (idx >= N_EDGES * NHEADS) return;
    int e = idx >> 2, h = idx & 3;
    int s = src[e], d = dst[e];
    const float* qp = qh + (size_t)d * HDIM + h * CDIM;
    const float* kp = kh + (size_t)s * HDIM + h * CDIM;
    const float* ep = eh + (size_t)e * HDIM + h * CDIM;
    float acc = 0.0f;
#pragma unroll
    for (int c = 0; c < CDIM; ++c) acc += qp[c] * (kp[c] + ep[c]);
    acc *= 0.17677669529663689f;   // 1/sqrt(32)
    logits[idx] = acc;
    atomicMaxF(&mbuf[d * NHEADS + h], acc);
}

__global__ void attn_exp_kernel(float* __restrict__ logits, const float* __restrict__ mbuf,
                                float* __restrict__ sbuf, const int* __restrict__ dst) {
    int idx = blockIdx.x * 256 + threadIdx.x;
    if (idx >= N_EDGES * NHEADS) return;
    int e = idx >> 2, h = idx & 3;
    int d = dst[e];
    float m = mbuf[d * NHEADS + h];
    if (!isfinite(m)) m = 0.0f;
    float ev = __expf(logits[idx] - m);
    logits[idx] = ev;
    atomicAdd(&sbuf[d * NHEADS + h], ev);
}

// dh[dst] += alpha * (vh[src] + eh[e]) over all channels
__global__ void attn_scatter_kernel(const float* __restrict__ logits, const float* __restrict__ sbuf,
                                    const float* __restrict__ vh, const float* __restrict__ eh,
                                    const int* __restrict__ src, const int* __restrict__ dst,
                                    float* __restrict__ dh) {
    int idx = blockIdx.x * 256 + threadIdx.x;
    if (idx >= N_EDGES * HDIM) return;
    int e = idx >> 7, ch = idx & 127;
    int h = ch >> 5;
    int d = dst[e];
    float alpha = logits[e * NHEADS + h] / (sbuf[d * NHEADS + h] + 1e-16f);
    float v = vh[(size_t)src[e] * HDIM + ch] + eh[(size_t)e * HDIM + ch];
    atomicAdd(&dh[(size_t)d * HDIM + ch], alpha * v);
}

// hV = LayerNorm(hV + add) * g + b  (one node per 128-thread block)
__global__ void ln_residual_kernel(float* __restrict__ hV, const float* __restrict__ add,
                                   const float* __restrict__ g, const float* __restrict__ b) {
    __shared__ float red[HDIM];
    int node = blockIdx.x, c = threadIdx.x;
    size_t off = (size_t)node * HDIM + c;
    float v = hV[off] + add[off];
    red[c] = v; __syncthreads();
    for (int s = HDIM / 2; s > 0; s >>= 1) { if (c < s) red[c] += red[c + s]; __syncthreads(); }
    float mean = red[0] * (1.0f / HDIM);
    __syncthreads();
    float d = v - mean;
    red[c] = d * d; __syncthreads();
    for (int s = HDIM / 2; s > 0; s >>= 1) { if (c < s) red[c] += red[c + s]; __syncthreads(); }
    float var = red[0] * (1.0f / HDIM);
    hV[off] = d * rsqrtf(var + 1e-5f) * g[c] + b[c];
}

__global__ void segsum_kernel(const float* __restrict__ hV, const int* __restrict__ batch,
                              float* __restrict__ sums, float* __restrict__ cnt) {
    int idx = blockIdx.x * 256 + threadIdx.x;
    if (idx >= N_NODES * HDIM) return;
    int node = idx >> 7, c = idx & 127;
    int g = batch[node];
    atomicAdd(&sums[g * HDIM + c], hV[idx]);
    if (c == 0) atomicAdd(&cnt[g], 1.0f);
}

__global__ void segdiv_kernel(const float* __restrict__ sums, const float* __restrict__ cnt,
                              float* __restrict__ out, int n) {
    int i = blockIdx.x * 256 + threadIdx.x;
    if (i >= n) return;
    out[i] = sums[i] / fmaxf(cnt[i >> 7], 1.0f);
}

__global__ void gate_apply_kernel(float* __restrict__ hV, const float* __restrict__ gate,
                                  const int* __restrict__ batch) {
    int idx = blockIdx.x * 256 + threadIdx.x;
    if (idx >= N_NODES * HDIM) return;
    int node = idx >> 7, c = idx & 127;
    hV[idx] *= gate[batch[node] * HDIM + c];
}

extern "C" void kernel_launch(void* const* d_in, const int* in_sizes, int n_in,
                              void* d_out, int out_size, void* d_ws, size_t ws_size,
                              hipStream_t stream) {
    (void)in_sizes; (void)n_in; (void)out_size; (void)ws_size;
    const float* x         = (const float*)d_in[0];
    const int*   eidx      = (const int*)d_in[1];
    const float* edge_attr = (const float*)d_in[2];
    const int*   batch_id  = (const int*)d_in[3];
    const int* src = eidx;
    const int* dst = eidx + N_EDGES;
    auto F = [&](int i) { return (const float*)d_in[i]; };
    // params flattened in setup_inputs() insertion order
    const float *Wnp=F(4), *bnp=F(5), *Wep=F(6), *bep=F(7), *bn_g=F(8), *bn_b=F(9);
    const float *Wv0=F(10), *bv0=F(11), *We0=F(12), *be0=F(13);
    const float bnMul = 1.0f / sqrtf(1.0f + 1e-5f);

    // ---- workspace layout (floats) ----
    float* ws = (float*)d_ws;
    float* hV   = ws;                 ws += (size_t)N_NODES * HDIM;
    float* dh   = ws;                 ws += (size_t)N_NODES * HDIM;
    float* qb   = ws;                 ws += (size_t)N_NODES * HDIM;
    float* kb   = ws;                 ws += (size_t)N_NODES * HDIM;
    float* vb   = ws;                 ws += (size_t)N_NODES * HDIM;
    float* ffb  = ws;                 ws += (size_t)N_NODES * 4 * HDIM;
    float* hE   = ws;                 ws += (size_t)N_EDGES * HDIM;
    float* ehb  = ws;                 ws += (size_t)N_EDGES * HDIM;   // eh / msg_pre
    float* lgt  = ws;                 ws += (size_t)N_EDGES * NHEADS;
    float* mbuf = ws;                 ws += (size_t)N_NODES * NHEADS;
    float* sbuf = ws;                 ws += (size_t)N_NODES * NHEADS;
    float* csum = ws;                 ws += (size_t)N_GRAPH * HDIM;   // cV
    float* ccnt = ws;                 ws += (size_t)N_GRAPH;
    float* gbuf = ws;                 ws += (size_t)N_GRAPH * HDIM;   // gate temp

    const dim3 blk(256);
    const dim3 gN(N_NODES / TM, 1);       // 128 blocks
    const dim3 gE(N_EDGES / TM, 1);       // 2048 blocks
    const dim3 gN4(N_NODES / TM, 4);      // FFN expand (Ntot=512)
    const dim3 gG(1, 1);                  // graph-level GEMMs (M=64)

    // ---- input projections (BN folded into epilogue) ----
    // node: ffb64 = bn(x @ Wnp + bnp) ; hV = ffb64 @ Wv0 + bv0
    gemm_wmma_kernel<ACT_NONE,0><<<gN, blk, 0, stream>>>(x, HDIM, Wnp, 64, bnp, nullptr,
        bn_g, bn_b, bnMul, ffb, 64, N_NODES, HDIM, 64, nullptr, nullptr, nullptr, nullptr);
    gemm_wmma_kernel<ACT_NONE,0><<<gN, blk, 0, stream>>>(ffb, 64, Wv0, HDIM, bv0, nullptr,
        nullptr, nullptr, 1.0f, hV, HDIM, N_NODES, 64, HDIM, nullptr, nullptr, nullptr, nullptr);
    // edge: ehb64 = bn(edge_attr @ Wep + bep) ; hE = ehb64 @ We0 + be0
    gemm_wmma_kernel<ACT_NONE,0><<<gE, blk, 0, stream>>>(edge_attr, HDIM, Wep, 64, bep, nullptr,
        bn_g, bn_b, bnMul, ehb, 64, N_EDGES, HDIM, 64, nullptr, nullptr, nullptr, nullptr);
    gemm_wmma_kernel<ACT_NONE,0><<<gE, blk, 0, stream>>>(ehb, 64, We0, HDIM, be0, nullptr,
        nullptr, nullptr, 1.0f, hE, HDIM, N_EDGES, 64, HDIM, nullptr, nullptr, nullptr, nullptr);

    for (int l = 0; l < NLAYERS; ++l) {
        int pb = 14 + 26 * l;
        const float *Wq=F(pb+0), *bq=F(pb+1), *Wk=F(pb+2), *bk=F(pb+3);
        const float *Wv=F(pb+4), *bv=F(pb+5), *We=F(pb+6), *be=F(pb+7);
        const float *ln1_g=F(pb+8), *ln1_b=F(pb+9), *ln2_g=F(pb+10), *ln2_b=F(pb+11);
        const float *Wf1=F(pb+12), *bf1=F(pb+13), *Wf2=F(pb+14), *bf2=F(pb+15);
        const float *W11=F(pb+16), *b11=F(pb+17), *W12=F(pb+18), *b12=F(pb+19);
        const float *bne_g=F(pb+20), *bne_b=F(pb+21);
        const float *Wg1=F(pb+22), *bg1=F(pb+23), *Wg2=F(pb+24), *bg2=F(pb+25);

        // --- attention projections ---
        gemm_wmma_kernel<ACT_NONE,0><<<gN, blk, 0, stream>>>(hV, HDIM, Wq, HDIM, bq, nullptr,
            nullptr, nullptr, 1.0f, qb, HDIM, N_NODES, HDIM, HDIM, nullptr, nullptr, nullptr, nullptr);
        gemm_wmma_kernel<ACT_NONE,0><<<gN, blk, 0, stream>>>(hV, HDIM, Wk, HDIM, bk, nullptr,
            nullptr, nullptr, 1.0f, kb, HDIM, N_NODES, HDIM, HDIM, nullptr, nullptr, nullptr, nullptr);
        gemm_wmma_kernel<ACT_NONE,0><<<gN, blk, 0, stream>>>(hV, HDIM, Wv, HDIM, bv, nullptr,
            nullptr, nullptr, 1.0f, vb, HDIM, N_NODES, HDIM, HDIM, nullptr, nullptr, nullptr, nullptr);
        gemm_wmma_kernel<ACT_NONE,0><<<gE, blk, 0, stream>>>(hE, HDIM, We, HDIM, be, nullptr,
            nullptr, nullptr, 1.0f, ehb, HDIM, N_EDGES, HDIM, HDIM, nullptr, nullptr, nullptr, nullptr);

        // --- segment softmax attention ---
        fill_kernel<<<(N_NODES * NHEADS + 255) / 256, blk, 0, stream>>>(mbuf, -INFINITY, N_NODES * NHEADS);
        hipMemsetAsync(sbuf, 0, sizeof(float) * N_NODES * NHEADS, stream);
        hipMemsetAsync(dh, 0, sizeof(float) * N_NODES * HDIM, stream);
        attn_logits_kernel<<<(N_EDGES * NHEADS + 255) / 256, blk, 0, stream>>>(qb, kb, ehb, src, dst, lgt, mbuf);
        attn_exp_kernel<<<(N_EDGES * NHEADS + 255) / 256, blk, 0, stream>>>(lgt, mbuf, sbuf, dst);
        attn_scatter_kernel<<<(N_EDGES * HDIM + 255) / 256, blk, 0, stream>>>(lgt, sbuf, vb, ehb, src, dst, dh);
        ln_residual_kernel<<<N_NODES, dim3(HDIM), 0, stream>>>(hV, dh, ln1_g, ln1_b);

        // --- FFN ---
        gemm_wmma_kernel<ACT_RELU,0><<<gN4, blk, 0, stream>>>(hV, HDIM, Wf1, 4 * HDIM, bf1, nullptr,
            nullptr, nullptr, 1.0f, ffb, 4 * HDIM, N_NODES, HDIM, 4 * HDIM, nullptr, nullptr, nullptr, nullptr);
        gemm_wmma_kernel<ACT_NONE,0><<<gN, blk, 0, stream>>>(ffb, 4 * HDIM, Wf2, HDIM, bf2, nullptr,
            nullptr, nullptr, 1.0f, dh, HDIM, N_NODES, 4 * HDIM, HDIM, nullptr, nullptr, nullptr, nullptr);
        ln_residual_kernel<<<N_NODES, dim3(HDIM), 0, stream>>>(hV, dh, ln2_g, ln2_b);

        // --- EdgeMLP: gather-concat GEMM (K=384) -> gelu -> GEMM -> +hE -> BN ---
        gemm_wmma_kernel<ACT_GELU,1><<<gE, blk, 0, stream>>>(nullptr, 0, W11, HDIM, b11, nullptr,
            nullptr, nullptr, 1.0f, ehb, HDIM, N_EDGES, 3 * HDIM, HDIM, src, dst, hV, hE);
        gemm_wmma_kernel<ACT_NONE,0><<<gE, blk, 0, stream>>>(ehb, HDIM, W12, HDIM, b12, hE,
            bne_g, bne_b, bnMul, hE, HDIM, N_EDGES, HDIM, HDIM, nullptr, nullptr, nullptr, nullptr);

        // --- context gate ---
        hipMemsetAsync(csum, 0, sizeof(float) * N_GRAPH * HDIM, stream);
        hipMemsetAsync(ccnt, 0, sizeof(float) * N_GRAPH, stream);
        segsum_kernel<<<(N_NODES * HDIM + 255) / 256, blk, 0, stream>>>(hV, batch_id, csum, ccnt);
        segdiv_kernel<<<(N_GRAPH * HDIM + 255) / 256, blk, 0, stream>>>(csum, ccnt, csum, N_GRAPH * HDIM);
        gemm_wmma_kernel<ACT_RELU,0><<<gG, blk, 0, stream>>>(csum, HDIM, Wg1, HDIM, bg1, nullptr,
            nullptr, nullptr, 1.0f, gbuf, HDIM, N_GRAPH, HDIM, HDIM, nullptr, nullptr, nullptr, nullptr);
        gemm_wmma_kernel<ACT_SIGMOID,0><<<gG, blk, 0, stream>>>(gbuf, HDIM, Wg2, HDIM, bg2, nullptr,
            nullptr, nullptr, 1.0f, csum, HDIM, N_GRAPH, HDIM, HDIM, nullptr, nullptr, nullptr, nullptr);
        gate_apply_kernel<<<(N_NODES * HDIM + 255) / 256, blk, 0, stream>>>(hV, csum, batch_id);
    }

    // ---- global mean pool -> d_out [G,1,H] ----
    hipMemsetAsync(csum, 0, sizeof(float) * N_GRAPH * HDIM, stream);
    hipMemsetAsync(ccnt, 0, sizeof(float) * N_GRAPH, stream);
    segsum_kernel<<<(N_NODES * HDIM + 255) / 256, blk, 0, stream>>>(hV, batch_id, csum, ccnt);
    segdiv_kernel<<<(N_GRAPH * HDIM + 255) / 256, blk, 0, stream>>>(csum, ccnt, (float*)d_out, N_GRAPH * HDIM);
}
